// MyMultiheadAttention_74646531604978
// MI455X (gfx1250) — matrix operand order repaired and checked by reading, not compiled, not run on hardware
//
#include <hip/hip_runtime.h>
#include <hip/hip_bf16.h>
#include <math.h>

// ---------------------------------------------------------------------------
// MI455X (gfx1250) multi-head attention forward, bf16 WMMA with f32 accum.
// S=2048, B=2, E=1024, H=16, D=64.  M = S*B = 4096 rows for all projections.
// Register-lean restructuring: B-fragments streamed one column at a time to
// keep peak live VGPRs < 256 (no scratch spills in the WMMA inner loops).
// ---------------------------------------------------------------------------

#define S_LEN 2048
#define BATCH 2
#define EMB   1024
#define NH    16
#define HD    64
#define MROWS (S_LEN * BATCH) // 4096

typedef __attribute__((ext_vector_type(8)))  __bf16 v8bf;
typedef __attribute__((ext_vector_type(16))) __bf16 v16bf;
typedef __attribute__((ext_vector_type(8)))  float  v8f;

static __device__ __forceinline__ v8f zero8() {
  v8f z;
#pragma unroll
  for (int i = 0; i < 8; ++i) z[i] = 0.0f;
  return z;
}

// WMMA D = A(16x32 bf16) * B(32x16 bf16) + C(16x16 f32)
static __device__ __forceinline__ v8f wmma_bf16(v16bf a, v16bf b, v8f c) {
  return __builtin_amdgcn_wmma_f32_16x16x32_bf16(
      /*neg_a=*/false, a, /*neg_b=*/false, b,
      /*c_mod=*/(short)0, c, /*reuse_a=*/false, /*reuse_b=*/false);
}

// A-fragment (16x32, bf16): lane<16 holds K {0..7,16..23}, lane>=16 {8..15,24..31}
static __device__ __forceinline__ v16bf loadA_bf16(const __bf16* rowk0, int laneHalf) {
  const v8bf lo = *reinterpret_cast<const v8bf*>(rowk0 + laneHalf * 8);
  const v8bf hi = *reinterpret_cast<const v8bf*>(rowk0 + 16 + laneHalf * 8);
  v16bf r;
#pragma unroll
  for (int j = 0; j < 8; ++j) { r[j] = lo[j]; r[j + 8] = hi[j]; }
  return r;
}

// Same A-fragment but converting from f32 source on the fly.
static __device__ __forceinline__ v16bf loadA_f32(const float* rowk0, int laneHalf) {
  const v8f lo = *reinterpret_cast<const v8f*>(rowk0 + laneHalf * 8);
  const v8f hi = *reinterpret_cast<const v8f*>(rowk0 + 16 + laneHalf * 8);
  v16bf r;
#pragma unroll
  for (int j = 0; j < 8; ++j) { r[j] = (__bf16)lo[j]; r[j + 8] = (__bf16)hi[j]; }
  return r;
}

// B-fragment (32x16, bf16): lane<16 holds K {0..7,16..23}, lane>=16 {8..15,24..31}
static __device__ __forceinline__ v16bf loadB_bf16(const __bf16* colk0, int laneHalf) {
  const v8bf lo = *reinterpret_cast<const v8bf*>(colk0 + laneHalf * 8);
  const v8bf hi = *reinterpret_cast<const v8bf*>(colk0 + 16 + laneHalf * 8);
  v16bf r;
#pragma unroll
  for (int j = 0; j < 8; ++j) { r[j] = lo[j]; r[j + 8] = hi[j]; }
  return r;
}

// ---------------------------------------------------------------------------
// f32 -> bf16 conversion (weights)
// ---------------------------------------------------------------------------
__global__ void f32_to_bf16_kernel(const float* __restrict__ in,
                                   __bf16* __restrict__ out, int n) {
  int i = blockIdx.x * blockDim.x + threadIdx.x;
  if (i < n) out[i] = (__bf16)in[i];
}

// ---------------------------------------------------------------------------
// GEMM: Out[M,N] = alpha * (A[M,K] @ W[N,K]^T + bias[N])
// Block = 128 threads = 4 waves, block tile 128x128, wave tile 64x64.
// B-fragments streamed per output column group: peak live = 128 acc + 32 A
// + 8 B + addressing.
// ---------------------------------------------------------------------------
template <bool A_IS_F32, bool OUT_BF16>
__global__ void __launch_bounds__(128, 1)
gemm_bias_kernel(const void* __restrict__ Av,
                 const __bf16* __restrict__ W,
                 const float* __restrict__ bias,
                 void* __restrict__ Out,
                 int Ndim, int Kdim, float alpha) {
  const int lane     = threadIdx.x & 31;
  const int wid      = threadIdx.x >> 5;
  const int laneM    = lane & 15;
  const int laneHalf = lane >> 4;
  const int n0 = blockIdx.x * 128 + (wid >> 1) * 64;
  const int m0 = blockIdx.y * 128 + (wid & 1) * 64;

  v8f acc[4][4];
#pragma unroll
  for (int i = 0; i < 4; ++i)
#pragma unroll
    for (int j = 0; j < 4; ++j) acc[i][j] = zero8();

  for (int k0 = 0; k0 < Kdim; k0 += 32) {
    v16bf af[4];
#pragma unroll
    for (int mi = 0; mi < 4; ++mi) {
      const int row = m0 + mi * 16 + laneM;
      if constexpr (A_IS_F32) {
        const float* p = (const float*)Av + (size_t)row * Kdim + k0;
        af[mi] = loadA_f32(p, laneHalf);
        __builtin_prefetch((const void*)(p + 32), 0, 1);
      } else {
        const __bf16* p = (const __bf16*)Av + (size_t)row * Kdim + k0;
        af[mi] = loadA_bf16(p, laneHalf);
        __builtin_prefetch((const void*)(p + 32), 0, 1);
      }
    }
#pragma unroll
    for (int ni = 0; ni < 4; ++ni) {
      const int col = n0 + ni * 16 + laneM;
      const __bf16* p = W + (size_t)col * Kdim + k0;
      const v16bf bfrag = loadB_bf16(p, laneHalf);
      __builtin_prefetch((const void*)(p + 32), 0, 1);
#pragma unroll
      for (int mi = 0; mi < 4; ++mi)
        acc[mi][ni] = wmma_bf16(af[mi], bfrag, acc[mi][ni]);
    }
  }

  // Epilogue: C layout -> row = m0+mi*16+laneHalf*8+r, col = n0+ni*16+laneM
#pragma unroll
  for (int ni = 0; ni < 4; ++ni) {
    const int col = n0 + ni * 16 + laneM;
    const float bv = bias[col];
#pragma unroll
    for (int mi = 0; mi < 4; ++mi) {
      const int rbase = m0 + mi * 16 + laneHalf * 8;
#pragma unroll
      for (int r = 0; r < 8; ++r) {
        const float v = (acc[mi][ni][r] + bv) * alpha;
        if constexpr (OUT_BF16)
          ((__bf16*)Out)[(size_t)(rbase + r) * Ndim + col] = (__bf16)v;
        else
          ((float*)Out)[(size_t)(rbase + r) * Ndim + col] = v;
      }
    }
  }
}

// ---------------------------------------------------------------------------
// Vt[b][h][d][s] = V[(s*B+b)*E + h*64 + d]   (bf16), so P@V B-fragments are
// contiguous along the contraction dimension (s).
// ---------------------------------------------------------------------------
__global__ void v_transpose_kernel(const __bf16* __restrict__ V,
                                   __bf16* __restrict__ Vt) {
  int i = blockIdx.x * blockDim.x + threadIdx.x; // B*H*D*S = 4M
  int s = i & (S_LEN - 1);
  int d = (i >> 11) & (HD - 1);
  int h = (i >> 17) & (NH - 1);
  int b = (i >> 21) & 1;
  Vt[i] = V[(size_t)(s * BATCH + b) * EMB + h * HD + d];
}

// ---------------------------------------------------------------------------
// Pass 1: exact per-row softmax stats (max, 1/sumexp) for each (b,h,row).
// One wave handles a 32-row q tile; scores computed via WMMA, reduced with
// width-16 shuffles across the 16 lanes that hold distinct columns.
// ---------------------------------------------------------------------------
__global__ void __launch_bounds__(32, 1)
attn_stats_kernel(const __bf16* __restrict__ Q,
                  const __bf16* __restrict__ Kb,
                  float* __restrict__ Mrow,
                  float* __restrict__ Linv) {
  const int lane = threadIdx.x & 31;
  const int laneM = lane & 15, laneHalf = lane >> 4;
  const int q0 = blockIdx.x * 32;
  const int bh = blockIdx.y;
  const int b = bh >> 4, h = bh & 15;

  v16bf qf[2][2];
#pragma unroll
  for (int mi = 0; mi < 2; ++mi) {
    const int srow = q0 + mi * 16 + laneM;
    const __bf16* p = Q + (size_t)(srow * BATCH + b) * EMB + h * HD;
#pragma unroll
    for (int ks = 0; ks < 2; ++ks) qf[mi][ks] = loadA_bf16(p + ks * 32, laneHalf);
  }

  float rm[2][8], rl[2][8];
#pragma unroll
  for (int mi = 0; mi < 2; ++mi)
#pragma unroll
    for (int r = 0; r < 8; ++r) { rm[mi][r] = -3.0e38f; rl[mi][r] = 0.0f; }

  for (int kt = 0; kt < S_LEN; kt += 64) {
    v8f acc[2][4];
#pragma unroll
    for (int mi = 0; mi < 2; ++mi)
#pragma unroll
      for (int ni = 0; ni < 4; ++ni) acc[mi][ni] = zero8();

#pragma unroll
    for (int ni = 0; ni < 4; ++ni) {
      const int krow = kt + ni * 16 + laneM;
      const __bf16* p = Kb + (size_t)(krow * BATCH + b) * EMB + h * HD;
#pragma unroll
      for (int ks = 0; ks < 2; ++ks) {
        const v16bf kf = loadB_bf16(p + ks * 32, laneHalf);
#pragma unroll
        for (int mi = 0; mi < 2; ++mi)
          acc[mi][ni] = wmma_bf16(qf[mi][ks], kf, acc[mi][ni]);
      }
    }

#pragma unroll
    for (int mi = 0; mi < 2; ++mi)
#pragma unroll
      for (int r = 0; r < 8; ++r) {
        float tmax = acc[mi][0][r];
#pragma unroll
        for (int ni = 1; ni < 4; ++ni) tmax = fmaxf(tmax, acc[mi][ni][r]);
#pragma unroll
        for (int off = 8; off >= 1; off >>= 1)
          tmax = fmaxf(tmax, __shfl_xor(tmax, off, 16));
        float tsum = 0.0f;
#pragma unroll
        for (int ni = 0; ni < 4; ++ni) tsum += __expf(acc[mi][ni][r] - tmax);
#pragma unroll
        for (int off = 8; off >= 1; off >>= 1)
          tsum += __shfl_xor(tsum, off, 16);
        const float nm = fmaxf(rm[mi][r], tmax);
        rl[mi][r] = rl[mi][r] * __expf(rm[mi][r] - nm) + tsum * __expf(tmax - nm);
        rm[mi][r] = nm;
      }
  }

  if (laneM == 0) {
#pragma unroll
    for (int mi = 0; mi < 2; ++mi)
#pragma unroll
      for (int r = 0; r < 8; ++r) {
        const int row = q0 + mi * 16 + laneHalf * 8 + r;
        const size_t idx = (size_t)bh * S_LEN + row;
        Mrow[idx] = rm[mi][r];
        Linv[idx] = 1.0f / rl[mi][r];
      }
  }
}

// ---------------------------------------------------------------------------
// Pass 2: O = softmax(QK^T) @ V.  P tile goes through LDS to be re-laid-out as
// a WMMA A-fragment; V is consumed from Vt (k-contiguous B-fragments).
// Writes AO[(s*B+b)*E + h*64+d] as bf16 for the final projection.
// ---------------------------------------------------------------------------
__global__ void __launch_bounds__(32, 1)
attn_out_kernel(const __bf16* __restrict__ Q,
                const __bf16* __restrict__ Kb,
                const __bf16* __restrict__ Vt,
                const float* __restrict__ Mrow,
                const float* __restrict__ Linv,
                __bf16* __restrict__ AO) {
  __shared__ __align__(16) __bf16 pt[32 * 72]; // 32x64 P tile, padded stride 72
  const int lane = threadIdx.x & 31;
  const int laneM = lane & 15, laneHalf = lane >> 4;
  const int q0 = blockIdx.x * 32;
  const int bh = blockIdx.y;
  const int b = bh >> 4, h = bh & 15;

  v16bf qf[2][2];
#pragma unroll
  for (int mi = 0; mi < 2; ++mi) {
    const int srow = q0 + mi * 16 + laneM;
    const __bf16* p = Q + (size_t)(srow * BATCH + b) * EMB + h * HD;
#pragma unroll
    for (int ks = 0; ks < 2; ++ks) qf[mi][ks] = loadA_bf16(p + ks * 32, laneHalf);
  }

  float mrow[2][8], linv[2][8];
#pragma unroll
  for (int mi = 0; mi < 2; ++mi)
#pragma unroll
    for (int r = 0; r < 8; ++r) {
      const int row = q0 + mi * 16 + laneHalf * 8 + r;
      mrow[mi][r] = Mrow[(size_t)bh * S_LEN + row];
      linv[mi][r] = Linv[(size_t)bh * S_LEN + row];
    }

  v8f oacc[2][4];
#pragma unroll
  for (int mi = 0; mi < 2; ++mi)
#pragma unroll
    for (int ni = 0; ni < 4; ++ni) oacc[mi][ni] = zero8();

  for (int kt = 0; kt < S_LEN; kt += 64) {
    // scores tile (32 x 64), K-fragments streamed per column group
    v8f acc[2][4];
#pragma unroll
    for (int mi = 0; mi < 2; ++mi)
#pragma unroll
      for (int ni = 0; ni < 4; ++ni) acc[mi][ni] = zero8();

#pragma unroll
    for (int ni = 0; ni < 4; ++ni) {
      const int krow = kt + ni * 16 + laneM;
      const __bf16* p = Kb + (size_t)(krow * BATCH + b) * EMB + h * HD;
#pragma unroll
      for (int ks = 0; ks < 2; ++ks) {
        const v16bf kf = loadB_bf16(p + ks * 32, laneHalf);
#pragma unroll
        for (int mi = 0; mi < 2; ++mi)
          acc[mi][ni] = wmma_bf16(qf[mi][ks], kf, acc[mi][ni]);
      }
    }

    // exact probabilities -> LDS (C layout -> row-major tile)
#pragma unroll
    for (int mi = 0; mi < 2; ++mi)
#pragma unroll
      for (int ni = 0; ni < 4; ++ni)
#pragma unroll
        for (int r = 0; r < 8; ++r) {
          const float pv = __expf(acc[mi][ni][r] - mrow[mi][r]) * linv[mi][r];
          pt[(mi * 16 + laneHalf * 8 + r) * 72 + ni * 16 + laneM] = (__bf16)pv;
        }
    __syncthreads();

    // P as A-fragments from LDS
    v16bf pf[2][2];
#pragma unroll
    for (int mi = 0; mi < 2; ++mi) {
      const int prow = mi * 16 + laneM;
#pragma unroll
      for (int ks = 0; ks < 2; ++ks)
        pf[mi][ks] = loadA_bf16(&pt[prow * 72 + ks * 32], laneHalf);
    }

    // V B-fragments from Vt, streamed per d column group
#pragma unroll
    for (int ni = 0; ni < 4; ++ni) {
      const int dcol = ni * 16 + laneM;
      const __bf16* vp = Vt + ((size_t)bh * HD + dcol) * S_LEN + kt;
#pragma unroll
      for (int ks = 0; ks < 2; ++ks) {
        const v16bf vf = loadB_bf16(vp + ks * 32, laneHalf);
#pragma unroll
        for (int mi = 0; mi < 2; ++mi)
          oacc[mi][ni] = wmma_bf16(pf[mi][ks], vf, oacc[mi][ni]);
      }
    }
    __syncthreads();
  }

#pragma unroll
  for (int mi = 0; mi < 2; ++mi)
#pragma unroll
    for (int ni = 0; ni < 4; ++ni)
#pragma unroll
      for (int r = 0; r < 8; ++r) {
        const int srow = q0 + mi * 16 + laneHalf * 8 + r;
        const int d = ni * 16 + laneM;
        AO[(size_t)(srow * BATCH + b) * EMB + h * HD + d] = (__bf16)oacc[mi][ni][r];
      }
}

// ---------------------------------------------------------------------------
// attn_avg[b][q][k] = (1/H) sum_h P_h[q][k].  Recompute scores per head with
// WMMA using the exact stats; deterministic (no atomics).
// ---------------------------------------------------------------------------
__global__ void __launch_bounds__(32, 1)
attn_avg_kernel(const __bf16* __restrict__ Q,
                const __bf16* __restrict__ Kb,
                const float* __restrict__ Mrow,
                const float* __restrict__ Linv,
                float* __restrict__ avg_out) {
  const int lane = threadIdx.x & 31;
  const int laneM = lane & 15, laneHalf = lane >> 4;
  const int kt = blockIdx.x * 64;
  const int q0 = blockIdx.y * 32;
  const int b = blockIdx.z;

  float avg[2][4][8];
#pragma unroll
  for (int mi = 0; mi < 2; ++mi)
#pragma unroll
    for (int ni = 0; ni < 4; ++ni)
#pragma unroll
      for (int r = 0; r < 8; ++r) avg[mi][ni][r] = 0.0f;

  for (int h = 0; h < NH; ++h) {
    const int bh = b * NH + h;
    v16bf qf[2][2];
#pragma unroll
    for (int mi = 0; mi < 2; ++mi) {
      const int srow = q0 + mi * 16 + laneM;
      const __bf16* p = Q + (size_t)(srow * BATCH + b) * EMB + h * HD;
#pragma unroll
      for (int ks = 0; ks < 2; ++ks) qf[mi][ks] = loadA_bf16(p + ks * 32, laneHalf);
    }

    v8f acc[2][4];
#pragma unroll
    for (int mi = 0; mi < 2; ++mi)
#pragma unroll
      for (int ni = 0; ni < 4; ++ni) acc[mi][ni] = zero8();

#pragma unroll
    for (int ni = 0; ni < 4; ++ni) {
      const int krow = kt + ni * 16 + laneM;
      const __bf16* p = Kb + (size_t)(krow * BATCH + b) * EMB + h * HD;
#pragma unroll
      for (int ks = 0; ks < 2; ++ks) {
        const v16bf kf = loadB_bf16(p + ks * 32, laneHalf);
#pragma unroll
        for (int mi = 0; mi < 2; ++mi)
          acc[mi][ni] = wmma_bf16(qf[mi][ks], kf, acc[mi][ni]);
      }
    }

#pragma unroll
    for (int mi = 0; mi < 2; ++mi)
#pragma unroll
      for (int r = 0; r < 8; ++r) {
        const int row = q0 + mi * 16 + laneHalf * 8 + r;
        const float m = Mrow[(size_t)bh * S_LEN + row];
        const float li = Linv[(size_t)bh * S_LEN + row];
#pragma unroll
        for (int ni = 0; ni < 4; ++ni)
          avg[mi][ni][r] += __expf(acc[mi][ni][r] - m) * li;
      }
  }

  const float sc = 1.0f / (float)NH;
#pragma unroll
  for (int mi = 0; mi < 2; ++mi)
#pragma unroll
    for (int ni = 0; ni < 4; ++ni)
#pragma unroll
      for (int r = 0; r < 8; ++r) {
        const int row = q0 + mi * 16 + laneHalf * 8 + r;
        const int col = kt + ni * 16 + laneM;
        avg_out[((size_t)b * S_LEN + row) * S_LEN + col] = avg[mi][ni][r] * sc;
      }
}

// ---------------------------------------------------------------------------
// Launch
// ---------------------------------------------------------------------------
extern "C" void kernel_launch(void* const* d_in, const int* in_sizes, int n_in,
                              void* d_out, int out_size, void* d_ws, size_t ws_size,
                              hipStream_t stream) {
  (void)in_sizes; (void)n_in; (void)out_size; (void)ws_size;
  const float* q_in = (const float*)d_in[0];
  const float* k_in = (const float*)d_in[1];
  const float* v_in = (const float*)d_in[2];
  const float* q_w  = (const float*)d_in[3];
  const float* q_b  = (const float*)d_in[4];
  const float* k_w  = (const float*)d_in[5];
  const float* k_b  = (const float*)d_in[6];
  const float* v_w  = (const float*)d_in[7];
  const float* v_b  = (const float*)d_in[8];
  const float* o_w  = (const float*)d_in[9];
  const float* o_b  = (const float*)d_in[10];

  float* Z = (float*)d_out;                          // [S,B,E]
  float* attn_avg = Z + (size_t)S_LEN * BATCH * EMB; // [B,S,S]

  char* ws = (char*)d_ws;
  size_t off = 0;
  auto alloc = [&](size_t bytes) -> char* {
    char* p = ws + off;
    off += (bytes + 255) & ~(size_t)255;
    return p;
  };
  __bf16* Wq = (__bf16*)alloc((size_t)EMB * EMB * 2);
  __bf16* Wk = (__bf16*)alloc((size_t)EMB * EMB * 2);
  __bf16* Wv = (__bf16*)alloc((size_t)EMB * EMB * 2);
  __bf16* Wo = (__bf16*)alloc((size_t)EMB * EMB * 2);
  __bf16* Qa = (__bf16*)alloc((size_t)MROWS * EMB * 2);
  __bf16* Ka = (__bf16*)alloc((size_t)MROWS * EMB * 2);
  __bf16* Va = (__bf16*)alloc((size_t)MROWS * EMB * 2);
  __bf16* Vt = (__bf16*)alloc((size_t)BATCH * NH * HD * S_LEN * 2);
  __bf16* AO = (__bf16*)alloc((size_t)MROWS * EMB * 2);
  float*  Mr = (float*)alloc((size_t)BATCH * NH * S_LEN * 4);
  float*  Li = (float*)alloc((size_t)BATCH * NH * S_LEN * 4);

  const int nw = EMB * EMB;
  f32_to_bf16_kernel<<<(nw + 255) / 256, 256, 0, stream>>>(q_w, Wq, nw);
  f32_to_bf16_kernel<<<(nw + 255) / 256, 256, 0, stream>>>(k_w, Wk, nw);
  f32_to_bf16_kernel<<<(nw + 255) / 256, 256, 0, stream>>>(v_w, Wv, nw);
  f32_to_bf16_kernel<<<(nw + 255) / 256, 256, 0, stream>>>(o_w, Wo, nw);

  const dim3 gblk(EMB / 128, MROWS / 128);
  const float scaling = 0.125f; // 1/sqrt(64)
  gemm_bias_kernel<true, true><<<gblk, 128, 0, stream>>>(q_in, Wq, q_b, Qa, EMB, EMB, scaling);
  gemm_bias_kernel<true, true><<<gblk, 128, 0, stream>>>(k_in, Wk, k_b, Ka, EMB, EMB, 1.0f);
  gemm_bias_kernel<true, true><<<gblk, 128, 0, stream>>>(v_in, Wv, v_b, Va, EMB, EMB, 1.0f);

  const int nt = BATCH * NH * HD * S_LEN;
  v_transpose_kernel<<<(nt + 255) / 256, 256, 0, stream>>>(Va, Vt);

  attn_stats_kernel<<<dim3(S_LEN / 32, BATCH * NH), 32, 0, stream>>>(Qa, Ka, Mr, Li);
  attn_out_kernel<<<dim3(S_LEN / 32, BATCH * NH), 32, 0, stream>>>(Qa, Ka, Vt, Mr, Li, AO);
  attn_avg_kernel<<<dim3(S_LEN / 64, S_LEN / 32, BATCH), 32, 0, stream>>>(Qa, Ka, Mr, Li, attn_avg);

  gemm_bias_kernel<false, false><<<gblk, 128, 0, stream>>>(AO, Wo, o_b, Z, EMB, EMB, 1.0f);
}
